// TransformerLayer_39805756899948
// MI455X (gfx1250) — matrix operand order repaired and checked by reading, not compiled, not run on hardware
//
#include <hip/hip_runtime.h>
#include <stdint.h>

#define D_MODEL 1024
#define N_HEADS 16
#define DK      64
#define D_FF    4096
#define BATCH   4
#define SEQ     2048
#define ROWS    (BATCH * SEQ) /* 8192 */

typedef __attribute__((ext_vector_type(16))) __bf16 v16bf;
typedef __attribute__((ext_vector_type(8)))  float  v8f;

union FragBF { v16bf v; uint4 q[2]; unsigned short u[16]; };
union Cvt8   { uint4 q;  unsigned short u[8]; };
union Cvt16  { uint4 q[2]; unsigned short u[16]; };

#ifndef __has_builtin
#define __has_builtin(x) 0
#endif

#if __has_builtin(__builtin_amdgcn_global_load_async_to_lds_b128)
#define HAVE_ASYNC_LDS 1
#else
#define HAVE_ASYNC_LDS 0
#endif

#if HAVE_ASYNC_LDS
// Builtin signature (from hipcc diagnostic): arg0 = AS1 pointer to
// vector_size(16) int, arg1 = AS3 pointer, then imm offset, imm cpol.
typedef int async_v4i __attribute__((vector_size(16)));
typedef __attribute__((address_space(1))) async_v4i* g_v4i_p;
typedef __attribute__((address_space(3))) async_v4i* l_v4i_p;

// per-lane 16B DMA: global -> LDS, tracked by ASYNCcnt (no VGPR round trip)
static __device__ __forceinline__ void async_copy16(const void* g, void* l) {
    __builtin_amdgcn_global_load_async_to_lds_b128(
        (g_v4i_p)(unsigned long long)(uintptr_t)g,
        (l_v4i_p)l, 0, 0);
}
static __device__ __forceinline__ void async_wait0() {
#if __has_builtin(__builtin_amdgcn_s_wait_asynccnt)
    __builtin_amdgcn_s_wait_asynccnt(0);
#else
    asm volatile("s_wait_asynccnt 0x0" ::: "memory");
#endif
}
#else
static __device__ __forceinline__ void async_wait0() {}
#endif

static __device__ __forceinline__ unsigned short f32_to_bf16(float f) {
    unsigned u = __float_as_uint(f);
    u += 0x7FFFu + ((u >> 16) & 1u);   // round-to-nearest-even
    return (unsigned short)(u >> 16);
}

__global__ __launch_bounds__(256) void cvt_f32_to_bf16_kernel(
    const float* __restrict__ in, unsigned short* __restrict__ out, int n) {
    int i = blockIdx.x * 256 + threadIdx.x;
    if (i < n) out[i] = f32_to_bf16(in[i]);
}

// ---------------------------------------------------------------------------
// Tiled bf16 WMMA GEMM, double-buffered LDS:  C = A @ B (+bias) (ReLU?)
// BM=128 (8 waves x 16 rows), BN=64 (4 n-tiles/wave), BK=32.
// bf16-A tiles are staged with GLOBAL_LOAD_ASYNC_TO_LDS_B128 (ASYNCcnt);
// f32-A tiles are converted in-flight; B tiles are transposed during staging
// so every WMMA fragment is a contiguous, 16B-aligned LDS read matching the
// documented wave32 A/B VGPR layouts.
// Batched via z: ofs = (z/Hdiv)*outer + (z%Hdiv)*inner  (Hdiv=1 -> unbatched)
// ---------------------------------------------------------------------------
#define BM 128
#define BN 64
#define BK 32
#define APITCH 40   // elems; rows 80B -> 16B-aligned accesses
#define BPITCH 40

template <int A_F32, int OUT_BF16, int RELU>
__global__ __launch_bounds__(256) void gemm_wmma_kernel(
    const void* __restrict__ Ag, const unsigned short* __restrict__ Bg,
    const float* __restrict__ bias, void* __restrict__ Cg,
    int M, int N, int K, int lda, int ldb, int ldc,
    int Hdiv,
    long long aOuter, long long aInner,
    long long bOuter, long long bInner,
    long long cOuter, long long cInner)
{
    (void)M; (void)N;
    __shared__ unsigned short lds_a[2][BM * APITCH];
    __shared__ unsigned short lds_b[2][BN * BPITCH];

    const int tid  = threadIdx.x;
    const int wid  = tid >> 5;
    const int lane = tid & 31;
    const int nl   = lane & 15;   // lane's M (A) / N (B,C) index
    const int hl   = lane >> 4;   // lane half

    const int z = blockIdx.z;
    const long long aofs = (long long)(z / Hdiv) * aOuter + (long long)(z % Hdiv) * aInner;
    const long long bofs = (long long)(z / Hdiv) * bOuter + (long long)(z % Hdiv) * bInner;
    const long long cofs = (long long)(z / Hdiv) * cOuter + (long long)(z % Hdiv) * cInner;

    const int mBlk = blockIdx.y * BM;
    const int nBlk = blockIdx.x * BN;

    const v8f vzero = {0.f, 0.f, 0.f, 0.f, 0.f, 0.f, 0.f, 0.f};
    v8f acc[4];
    for (int t = 0; t < 4; ++t) acc[t] = vzero;

    // staging coordinates
    const int ar  = tid >> 1;         // A row 0..127
    const int ah  = tid & 1;          // 16-elem half along K
    const int bk  = tid >> 3;         // B k 0..31
    const int bnq = (tid & 7) * 8;    // B n-quad base 0..56

    auto stageA = [&](int buf, int k0) {
        if (A_F32) {
            const float* A = (const float*)Ag + aofs;
            const float4* src = (const float4*)(A + (long long)(mBlk + ar) * lda + k0 + ah * 16);
            Cvt16 c16;
            #pragma unroll
            for (int q = 0; q < 4; ++q) {
                float4 f = src[q];
                c16.u[q * 4 + 0] = f32_to_bf16(f.x);
                c16.u[q * 4 + 1] = f32_to_bf16(f.y);
                c16.u[q * 4 + 2] = f32_to_bf16(f.z);
                c16.u[q * 4 + 3] = f32_to_bf16(f.w);
            }
            uint4* dst = (uint4*)&lds_a[buf][ar * APITCH + ah * 16];
            dst[0] = c16.q[0]; dst[1] = c16.q[1];
        } else {
            const unsigned short* A = (const unsigned short*)Ag + aofs;
            const unsigned short* g = A + (long long)(mBlk + ar) * lda + k0 + ah * 16;
            unsigned short* l = &lds_a[buf][ar * APITCH + ah * 16];
#if HAVE_ASYNC_LDS
            async_copy16(g, l);
            async_copy16(g + 8, l + 8);
#else
            uint4* dst = (uint4*)l;
            dst[0] = ((const uint4*)g)[0];
            dst[1] = ((const uint4*)g)[1];
#endif
        }
    };
    auto stageB = [&](int buf, int k0) {
        const unsigned short* B = Bg + bofs;
        const unsigned short* src = B + (long long)(k0 + bk) * ldb + nBlk + bnq;
        Cvt8 t8;
        t8.q = *(const uint4*)src;
        #pragma unroll
        for (int j = 0; j < 8; ++j)
            lds_b[buf][(bnq + j) * BPITCH + bk] = t8.u[j];
        if (k0 + BK < K)
            __builtin_prefetch((const char*)src + (size_t)BK * ldb * 2, 0, 1);
    };

    const int NS = K / BK;
    stageA(0, 0);
    stageB(0, 0);
    if (!A_F32) async_wait0();
    __syncthreads();

    for (int s = 0; s < NS; ++s) {
        const int cur = s & 1;
        if (s + 1 < NS) {                 // DMA next tile while computing
            stageA(1 - cur, (s + 1) * BK);
            stageB(1 - cur, (s + 1) * BK);
        }

        FragBF fa;
        const int am = wid * 16 + nl;
        fa.q[0] = *(const uint4*)&lds_a[cur][am * APITCH + 8 * hl];
        fa.q[1] = *(const uint4*)&lds_a[cur][am * APITCH + 16 + 8 * hl];
        #pragma unroll
        for (int t = 0; t < 4; ++t) {
            FragBF fb;
            fb.q[0] = *(const uint4*)&lds_b[cur][(t * 16 + nl) * BPITCH + 8 * hl];
            fb.q[1] = *(const uint4*)&lds_b[cur][(t * 16 + nl) * BPITCH + 16 + 8 * hl];
            acc[t] = __builtin_amdgcn_wmma_f32_16x16x32_bf16(
                false, fa.v, false, fb.v, (short)0, acc[t], false, false);
        }

        if (s + 1 < NS) {
            if (!A_F32) async_wait0();    // next tile's DMA must have landed
            __syncthreads();
        }
    }

    // ---- store C (f32 C/D layout: lane n=L%16, VGPR r -> row 8*hl+r) ----
    const int mw = mBlk + wid * 16;
    #pragma unroll
    for (int t = 0; t < 4; ++t) {
        #pragma unroll
        for (int r = 0; r < 8; ++r) {
            int row = mw + 8 * hl + r;
            int col = nBlk + t * 16 + nl;
            float vsum = acc[t][r];
            if (bias) vsum += bias[col];
            if (RELU) vsum = fmaxf(vsum, 0.0f);
            if (OUT_BF16)
                ((unsigned short*)Cg)[cofs + (long long)row * ldc + col] = f32_to_bf16(vsum);
            else
                ((float*)Cg)[cofs + (long long)row * ldc + col] = vsum;
        }
    }
}

// ---------------------------------------------------------------------------
// Attention scores: per-wave 16x16 tile of Q@K^T, scaled, distance-decayed,
// causally masked. q/k are bf16 [B*S, D_MODEL]; per (b,h) slice of 64 cols.
// K^T columns are K rows -> B fragments are two 16B global loads per lane (k0+8*hl, k0+16+8*hl).
// ---------------------------------------------------------------------------
__global__ __launch_bounds__(256) void attn_scores_kernel(
    const unsigned short* __restrict__ qb, const unsigned short* __restrict__ kb,
    const float* __restrict__ gammas, float* __restrict__ att)
{
    const int tid  = threadIdx.x;
    const int wid  = tid >> 5;
    const int lane = tid & 31;
    const int nl   = lane & 15;
    const int hl   = lane >> 4;

    const int z  = blockIdx.z;            // b*H + h
    const int b  = z >> 4;
    const int h  = z & 15;
    const int m0 = blockIdx.y * 16;
    const int n0 = (blockIdx.x * 8 + wid) * 16;

    float* out = att + (long long)z * SEQ * SEQ;

    if (n0 > m0 + 15) {                   // fully masked tile
        #pragma unroll
        for (int r = 0; r < 8; ++r) {
            int i = m0 + 8 * hl + r, j = n0 + nl;
            out[(long long)i * SEQ + j] = -1e32f;
        }
        return;
    }

    float g  = gammas[h];
    float sp = (g > 20.f) ? g : log1pf(expf(g));   // softplus(gamma)

    v8f acc = {0.f, 0.f, 0.f, 0.f, 0.f, 0.f, 0.f, 0.f};
    const long long qrow = (long long)(b * SEQ + m0 + nl) * D_MODEL + h * DK;
    const long long krow = (long long)(b * SEQ + n0 + nl) * D_MODEL + h * DK;
    #pragma unroll
    for (int k0 = 0; k0 < DK; k0 += 32) {
        FragBF fa, fb;
        fa.q[0] = *(const uint4*)(qb + qrow + k0 + 8 * hl);
        fa.q[1] = *(const uint4*)(qb + qrow + k0 + 16 + 8 * hl);
        fb.q[0] = *(const uint4*)(kb + krow + k0 + 8 * hl);
        fb.q[1] = *(const uint4*)(kb + krow + k0 + 16 + 8 * hl);
        acc = __builtin_amdgcn_wmma_f32_16x16x32_bf16(
            false, fa.v, false, fb.v, (short)0, acc, false, false);
    }

    #pragma unroll
    for (int r = 0; r < 8; ++r) {
        int i = m0 + 8 * hl + r, j = n0 + nl;
        float s   = acc[r] * 0.125f;                 // 1/sqrt(64)
        float pe  = fabsf((float)(i - j)) * (1.0f / 21.0f);
        float eff = expf(pe * sp);
        eff = fminf(fmaxf(eff, 1e-5f), 1e5f);
        s *= eff;
        if (j > i) s = -1e32f;
        out[(long long)i * SEQ + j] = s;
    }
}

// Row softmax, in place over [B*H*S, S]
__global__ __launch_bounds__(256) void softmax_kernel(float* __restrict__ att) {
    __shared__ float red[256];
    const long long row = blockIdx.x;
    float* p = att + row * SEQ;
    const int tid = threadIdx.x;
    float v[8];
    float mx = -3.4e38f;
    #pragma unroll
    for (int u = 0; u < 8; ++u) { v[u] = p[tid + u * 256]; mx = fmaxf(mx, v[u]); }
    red[tid] = mx; __syncthreads();
    for (int s = 128; s > 0; s >>= 1) { if (tid < s) red[tid] = fmaxf(red[tid], red[tid + s]); __syncthreads(); }
    mx = red[0]; __syncthreads();
    float sum = 0.f;
    #pragma unroll
    for (int u = 0; u < 8; ++u) { v[u] = expf(v[u] - mx); sum += v[u]; }
    red[tid] = sum; __syncthreads();
    for (int s = 128; s > 0; s >>= 1) { if (tid < s) red[tid] += red[tid + s]; __syncthreads(); }
    float inv = 1.0f / red[0];
    #pragma unroll
    for (int u = 0; u < 8; ++u) p[tid + u * 256] = v[u] * inv;
}

// LayerNorm(a+b): writes f32 (+ optional bf16 copy for next GEMM)
__global__ __launch_bounds__(256) void ln_kernel(
    const float* __restrict__ a, const float* __restrict__ b,
    const float* __restrict__ scale, const float* __restrict__ bias,
    float* __restrict__ of32, unsigned short* __restrict__ obf16)
{
    __shared__ float red[256];
    const long long row = blockIdx.x;
    const int tid = threadIdx.x;
    float v[4];
    float s = 0.f;
    #pragma unroll
    for (int u = 0; u < 4; ++u) {
        int c = tid + u * 256;
        v[u] = a[row * D_MODEL + c] + b[row * D_MODEL + c];
        s += v[u];
    }
    red[tid] = s; __syncthreads();
    for (int t = 128; t > 0; t >>= 1) { if (tid < t) red[tid] += red[tid + t]; __syncthreads(); }
    float mu = red[0] * (1.0f / D_MODEL); __syncthreads();
    s = 0.f;
    #pragma unroll
    for (int u = 0; u < 4; ++u) { float d = v[u] - mu; s += d * d; }
    red[tid] = s; __syncthreads();
    for (int t = 128; t > 0; t >>= 1) { if (tid < t) red[tid] += red[tid + t]; __syncthreads(); }
    float rs = rsqrtf(red[0] * (1.0f / D_MODEL) + 1e-5f);
    #pragma unroll
    for (int u = 0; u < 4; ++u) {
        int c = tid + u * 256;
        float y = (v[u] - mu) * rs * scale[c] + bias[c];
        of32[row * D_MODEL + c] = y;
        if (obf16) obf16[row * D_MODEL + c] = f32_to_bf16(y);
    }
}

// ---------------------------------------------------------------------------
extern "C" void kernel_launch(void* const* d_in, const int* in_sizes, int n_in,
                              void* d_out, int out_size, void* d_ws, size_t ws_size,
                              hipStream_t stream)
{
    (void)in_sizes; (void)n_in; (void)out_size; (void)ws_size;
    const float* query  = (const float*)d_in[0];
    const float* key    = (const float*)d_in[1];
    const float* values = (const float*)d_in[2];
    const float* Wk   = (const float*)d_in[3];
    const float* bk   = (const float*)d_in[4];
    const float* Wv   = (const float*)d_in[5];
    const float* bv   = (const float*)d_in[6];
    const float* Wout = (const float*)d_in[7];
    const float* bout = (const float*)d_in[8];
    const float* gam  = (const float*)d_in[9];
    const float* l1s  = (const float*)d_in[10];
    const float* l1b  = (const float*)d_in[11];
    const float* W1   = (const float*)d_in[12];
    const float* b1   = (const float*)d_in[13];
    const float* W2   = (const float*)d_in[14];
    const float* b2   = (const float*)d_in[15];
    const float* l2s  = (const float*)d_in[16];
    const float* l2b  = (const float*)d_in[17];

    float* xOut = (float*)d_out;                                   // [B,S,D]
    float* att  = xOut + (long long)ROWS * D_MODEL;                // [B,H,S,S]

    // workspace carve-out
    char* w = (char*)d_ws;
    auto alloc = [&](size_t bytes) -> void* {
        void* p = (void*)w; w += (bytes + 255) & ~(size_t)255; return p;
    };
    unsigned short* wkb   = (unsigned short*)alloc((size_t)D_MODEL * D_MODEL * 2);
    unsigned short* wvb   = (unsigned short*)alloc((size_t)D_MODEL * D_MODEL * 2);
    unsigned short* woutb = (unsigned short*)alloc((size_t)D_MODEL * D_MODEL * 2);
    unsigned short* w1b   = (unsigned short*)alloc((size_t)D_MODEL * D_FF * 2);
    unsigned short* w2b   = (unsigned short*)alloc((size_t)D_FF * D_MODEL * 2);
    unsigned short* qib   = (unsigned short*)alloc((size_t)ROWS * D_MODEL * 2); // bf16(query)
    unsigned short* kib   = (unsigned short*)alloc((size_t)ROWS * D_MODEL * 2); // bf16(key)
    unsigned short* vib   = (unsigned short*)alloc((size_t)ROWS * D_MODEL * 2); // bf16(values)
    unsigned short* qb    = (unsigned short*)alloc((size_t)ROWS * D_MODEL * 2);
    unsigned short* kb    = (unsigned short*)alloc((size_t)ROWS * D_MODEL * 2);
    unsigned short* vb    = (unsigned short*)alloc((size_t)ROWS * D_MODEL * 2);
    unsigned short* ccb   = (unsigned short*)alloc((size_t)ROWS * D_MODEL * 2);
    float*          aOutF = (float*)alloc((size_t)ROWS * D_MODEL * 4);
    float*          xF    = (float*)alloc((size_t)ROWS * D_MODEL * 4);
    unsigned short* xB    = (unsigned short*)alloc((size_t)ROWS * D_MODEL * 2);
    unsigned short* h1b   = (unsigned short*)alloc((size_t)ROWS * D_FF * 2);
    float*          y2F   = (float*)alloc((size_t)ROWS * D_MODEL * 4);

    // 1) weights + input activations -> bf16
    const int nW = D_MODEL * D_MODEL, nF = D_MODEL * D_FF, nX = ROWS * D_MODEL;
    cvt_f32_to_bf16_kernel<<<(nW + 255) / 256, 256, 0, stream>>>(Wk, wkb, nW);
    cvt_f32_to_bf16_kernel<<<(nW + 255) / 256, 256, 0, stream>>>(Wv, wvb, nW);
    cvt_f32_to_bf16_kernel<<<(nW + 255) / 256, 256, 0, stream>>>(Wout, woutb, nW);
    cvt_f32_to_bf16_kernel<<<(nF + 255) / 256, 256, 0, stream>>>(W1, w1b, nF);
    cvt_f32_to_bf16_kernel<<<(nF + 255) / 256, 256, 0, stream>>>(W2, w2b, nF);
    cvt_f32_to_bf16_kernel<<<(nX + 255) / 256, 256, 0, stream>>>(query,  qib, nX);
    cvt_f32_to_bf16_kernel<<<(nX + 255) / 256, 256, 0, stream>>>(key,    kib, nX);
    cvt_f32_to_bf16_kernel<<<(nX + 255) / 256, 256, 0, stream>>>(values, vib, nX);

    // 2) q/k/v projections (q and k share Wk, faithful to reference)
    dim3 gProj(D_MODEL / BN, ROWS / BM, 1);
    gemm_wmma_kernel<0, 1, 0><<<gProj, 256, 0, stream>>>(
        qib, wkb, bk, qb, ROWS, D_MODEL, D_MODEL, D_MODEL, D_MODEL, D_MODEL,
        1, 0, 0, 0, 0, 0, 0);
    gemm_wmma_kernel<0, 1, 0><<<gProj, 256, 0, stream>>>(
        kib, wkb, bk, kb, ROWS, D_MODEL, D_MODEL, D_MODEL, D_MODEL, D_MODEL,
        1, 0, 0, 0, 0, 0, 0);
    gemm_wmma_kernel<0, 1, 0><<<gProj, 256, 0, stream>>>(
        vib, wvb, bv, vb, ROWS, D_MODEL, D_MODEL, D_MODEL, D_MODEL, D_MODEL,
        1, 0, 0, 0, 0, 0, 0);

    // 3) scores (scaled, decayed, masked) -> att region
    dim3 gScore(SEQ / 128, SEQ / 16, BATCH * N_HEADS);
    attn_scores_kernel<<<gScore, 256, 0, stream>>>(qb, kb, gam, att);

    // 4) row softmax in place -> final att_weight output
    softmax_kernel<<<BATCH * N_HEADS * SEQ, 256, 0, stream>>>(att);

    // 5) ctx = att @ V per (b,h), written into concat layout [B,S,H*64] bf16
    dim3 gCtx(DK / BN, SEQ / BM, BATCH * N_HEADS);
    gemm_wmma_kernel<1, 1, 0><<<gCtx, 256, 0, stream>>>(
        att, vb, nullptr, ccb, SEQ, DK, SEQ, SEQ, D_MODEL, D_MODEL,
        N_HEADS,
        16LL * SEQ * SEQ, (long long)SEQ * SEQ,          // A: z*S*S
        (long long)SEQ * D_MODEL, (long long)DK,         // B: b*S*D + h*64
        (long long)SEQ * D_MODEL, (long long)DK);        // C: b*S*D + h*64

    // 6) output projection -> attn_out f32
    gemm_wmma_kernel<0, 0, 0><<<gProj, 256, 0, stream>>>(
        ccb, woutb, bout, aOutF, ROWS, D_MODEL, D_MODEL, D_MODEL, D_MODEL, D_MODEL,
        1, 0, 0, 0, 0, 0, 0);

    // 7) LN1(query + attn_out) -> x (f32 + bf16)
    ln_kernel<<<ROWS, 256, 0, stream>>>(query, aOutF, l1s, l1b, xF, xB);

    // 8) FFN1 with ReLU -> h1 bf16
    dim3 gF1(D_FF / BN, ROWS / BM, 1);
    gemm_wmma_kernel<0, 1, 1><<<gF1, 256, 0, stream>>>(
        xB, w1b, b1, h1b, ROWS, D_FF, D_MODEL, D_MODEL, D_FF, D_FF,
        1, 0, 0, 0, 0, 0, 0);

    // 9) FFN2 -> y2 f32
    dim3 gF2(D_MODEL / BN, ROWS / BM, 1);
    gemm_wmma_kernel<0, 0, 0><<<gF2, 256, 0, stream>>>(
        h1b, w2b, b2, y2F, ROWS, D_MODEL, D_FF, D_FF, D_MODEL, D_MODEL,
        1, 0, 0, 0, 0, 0, 0);

    // 10) LN2(x + y2) -> final x output
    ln_kernel<<<ROWS, 256, 0, stream>>>(xF, y2F, l2s, l2b, xOut, nullptr);
}
